// Net_52905407152430
// MI455X (gfx1250) — matrix operand order repaired and checked by reading, not compiled, not run on hardware
//
#include <hip/hip_runtime.h>
#include <hip/hip_bf16.h>
#include <cstdint>
#include <cstddef>

// ---------------------------------------------------------------------------
// GraphSAGE 2-layer forward for MI455X (gfx1250, wave32, WMMA bf16).
// GEMMs via v_wmma_f32_16x16x32_bf16. Mean-divide and f32->bf16 conversion
// fused into branch-free, vectorized LDS staging (b128 loads, pk_bf16 cvt,
// b128 LDS stores). Row/col guards hoisted to block-uniform branches.
// ---------------------------------------------------------------------------

typedef __bf16 bf16;
typedef __attribute__((ext_vector_type(16))) __bf16 v16bf;
typedef __attribute__((ext_vector_type(8)))  float  v8f;

#define NNODE 100000
#define NEDGE 400000
#define DIN   512
#define DHID  256
#define DOUT  129

// ------------------------------- utility -----------------------------------

__global__ void zero_f32(float* __restrict__ p, long n) {
  long i = (long)blockIdx.x * blockDim.x + threadIdx.x;
  long stride = (long)gridDim.x * blockDim.x;
  for (; i < n; i += stride) p[i] = 0.0f;
}

__global__ void degree_k(const long long* __restrict__ dst, float* __restrict__ deg, int E) {
  int e = blockIdx.x * blockDim.x + threadIdx.x;
  if (e < E) atomicAdd(&deg[(int)dst[e]], 1.0f);
}

// scatter-add x[src] rows into msg[dst]; 4 floats per thread (vectorized load)
__global__ void scatter_x(const long long* __restrict__ src, const long long* __restrict__ dst,
                          const float* __restrict__ x, float* __restrict__ msg, int E) {
  int gid = blockIdx.x * blockDim.x + threadIdx.x;
  int e = gid >> 7;            // DIN/4 = 128 threads per edge
  int c = (gid & 127) << 2;
  if (e >= E) return;
  int s = (int)src[e], d = (int)dst[e];
  const float4 v = *(const float4*)(x + (size_t)s * DIN + c);
  float* o = msg + (size_t)d * DIN + c;
  atomicAdd(o + 0, v.x);
  atomicAdd(o + 1, v.y);
  atomicAdd(o + 2, v.z);
  atomicAdd(o + 3, v.w);
}

// scatter-add h[src] (bf16) rows into msg2[dst] (f32)
__global__ void scatter_h(const long long* __restrict__ src, const long long* __restrict__ dst,
                          const bf16* __restrict__ h, float* __restrict__ msg, int E) {
  int gid = blockIdx.x * blockDim.x + threadIdx.x;
  int e = gid >> 6;            // DHID/4 = 64 threads per edge
  int c = (gid & 63) << 2;
  if (e >= E) return;
  int s = (int)src[e], d = (int)dst[e];
  const bf16* hp = h + (size_t)s * DHID + c;
  float* o = msg + (size_t)d * DHID + c;
  #pragma unroll
  for (int i = 0; i < 4; ++i) atomicAdd(o + i, (float)hp[i]);
}

// ------------------------- WMMA fragment helpers ----------------------------
// A fragment (16x32 bf16): lane = half*16 + r holds row r; VGPR i holds K pair:
//   i<4 : K = half*8 + 2i ;  i>=4 : K = 16 + half*8 + 2(i-4)
__device__ inline v16bf load_afrag(const bf16* __restrict__ rowp, int half) {
  union { unsigned u[8]; v16bf v; } au;
  #pragma unroll
  for (int i = 0; i < 8; ++i) {
    int kb = (i < 4) ? (half * 8 + i * 2) : (16 + half * 8 + (i - 4) * 2);
    au.u[i] = *(const unsigned*)(rowp + kb);
  }
  return au.v;
}
// B fragment (32x16 bf16) staged transposed in LDS as [col][k]:
// lane = half*16 + c holds col c; VGPR i holds K pair (same as A):
//   i<4 : K = half*8 + 2i ;  i>=4 : K = 16 + half*8 + 2(i-4)
__device__ inline v16bf load_bfrag(const bf16* __restrict__ colp, int half) {
  union { unsigned u[8]; v16bf v; } bu;
  #pragma unroll
  for (int i = 0; i < 8; ++i) {
    int kb = (i < 4) ? (half * 8 + i * 2) : (16 + half * 8 + (i - 4) * 2);
    bu.u[i] = *(const unsigned*)(colp + kb);
  }
  return bu.v;
}

// pack 16 f32 (scaled) -> 16 bf16 and store 32B to LDS (two b128 stores)
__device__ inline void cvt_store16(bf16* __restrict__ dst, const float4 f[4], float scale) {
  union { bf16 h[16]; uint4 q[2]; } pk;
  #pragma unroll
  for (int i = 0; i < 4; ++i) {
    pk.h[i * 4 + 0] = (bf16)(f[i].x * scale);
    pk.h[i * 4 + 1] = (bf16)(f[i].y * scale);
    pk.h[i * 4 + 2] = (bf16)(f[i].z * scale);
    pk.h[i * 4 + 3] = (bf16)(f[i].w * scale);
  }
  uint4* q = (uint4*)dst;
  q[0] = pk.q[0];
  q[1] = pk.q[1];
}

// ------------------------------ layer-1 GEMM --------------------------------
// h = relu( (msg/deg) @ Wl + b + x @ Wr ), done as single K = 2*512 loop.
// Block tile 128x64; 8 waves of 32x32 (2x2 wmma 16x16x32 bf16).
__launch_bounds__(256)
__global__ void sage_gemm1(const float* __restrict__ msg, const float* __restrict__ deg,
                           const float* __restrict__ x,
                           const float* __restrict__ Wl, const float* __restrict__ Wr,
                           const float* __restrict__ b, bf16* __restrict__ hout) {
  const int M = NNODE, K = DIN, Nout = DHID;
  __shared__ __align__(16) bf16 sA[128][32];
  __shared__ __align__(16) bf16 sB[64][32];   // transposed: [col][k]
  __shared__ float sRD[128];

  const int tid = threadIdx.x;
  const int m0 = blockIdx.x * 128;
  const int n0 = blockIdx.y * 64;

  if (tid < 128) {
    int r = m0 + tid;
    float dv = (r < M) ? deg[r] : 1.0f;
    sRD[tid] = 1.0f / fmaxf(dv, 1.0f);
  }
  __syncthreads();

  const int w = tid >> 5, lane = tid & 31;
  const int wm = (w & 3) * 32, wn = (w >> 2) * 32;
  const int half = lane >> 4, lr = lane & 15;

  // A-staging map: one thread = one (row, 16k) chunk; clamped row (branch-free)
  const int ar  = tid >> 1;
  const int aks = (tid & 1) * 16;
  const int arow  = m0 + ar;
  const int arowc = (arow < M) ? arow : (M - 1);
  const float ascale0 = sRD[ar];
  // B-staging map: one thread = 8 consecutive cols at one k
  const int bk = tid >> 3;
  const int bn = (tid & 7) * 8;

  v8f acc[2][2] = {};

  for (int kt = 0; kt < 2 * K; kt += 32) {
    const int phase = (kt >= K) ? 1 : 0;
    const int k0 = kt - phase * K;

    // ---- stage A (128x32): agg (phase 0) or x (phase 1), branch-free ----
    {
      const float* srcp = phase ? x : msg;
      const float  scl  = phase ? 1.0f : ascale0;
      const float* ap   = srcp + (size_t)arowc * K + k0 + aks;
      float4 f[4];
      #pragma unroll
      for (int i = 0; i < 4; ++i) f[i] = ((const float4*)ap)[i];
      if (k0 + 32 < K) __builtin_prefetch(ap + 32, 0, 0);
      cvt_store16(&sA[ar][aks], f, scl);
    }
    // ---- stage B (32x64) transposed; Nout=256 so always in-bounds ----
    {
      const float* W  = phase ? Wr : Wl;
      const float* wp = W + (size_t)(k0 + bk) * Nout + n0 + bn;
      float4 f0 = ((const float4*)wp)[0];
      float4 f1 = ((const float4*)wp)[1];
      bf16 hh[8];
      hh[0] = (bf16)f0.x; hh[1] = (bf16)f0.y; hh[2] = (bf16)f0.z; hh[3] = (bf16)f0.w;
      hh[4] = (bf16)f1.x; hh[5] = (bf16)f1.y; hh[6] = (bf16)f1.z; hh[7] = (bf16)f1.w;
      #pragma unroll
      for (int i = 0; i < 8; ++i) sB[bn + i][bk] = hh[i];
    }
    __syncthreads();

    v16bf af[2], bfv[2];
    #pragma unroll
    for (int t = 0; t < 2; ++t) {
      af[t]  = load_afrag(&sA[wm + t * 16 + lr][0], half);
      bfv[t] = load_bfrag(&sB[wn + t * 16 + lr][0], half);
    }
    #pragma unroll
    for (int ti = 0; ti < 2; ++ti)
      #pragma unroll
      for (int tj = 0; tj < 2; ++tj)
        acc[ti][tj] = __builtin_amdgcn_wmma_f32_16x16x32_bf16(
            false, af[ti], false, bfv[tj], (short)0, acc[ti][tj], false, false);
    __syncthreads();
  }

  // C/D layout: VGPR j -> row (j + 8*half), lane%16 -> col
  const bool fullM = (m0 + 128 <= M);
  #pragma unroll
  for (int ti = 0; ti < 2; ++ti) {
    #pragma unroll
    for (int tj = 0; tj < 2; ++tj) {
      int col = n0 + wn + tj * 16 + lr;
      float bias = b[col];
      #pragma unroll
      for (int j = 0; j < 8; ++j) {
        int row = m0 + wm + ti * 16 + j + half * 8;
        if (fullM || row < M) {
          float v = acc[ti][tj][j] + bias;
          v = fmaxf(v, 0.0f);                      // relu
          hout[(size_t)row * Nout + col] = (bf16)v;
        }
      }
    }
  }
}

// ------------------------------ layer-2 GEMM --------------------------------
// logits = (msg2/deg) @ W2l + b2 + h @ W2r ; f32 out, Nout=129 (grid padded 192)
__launch_bounds__(256)
__global__ void sage_gemm2(const float* __restrict__ msg, const float* __restrict__ deg,
                           const bf16* __restrict__ h,
                           const float* __restrict__ Wl, const float* __restrict__ Wr,
                           const float* __restrict__ b, float* __restrict__ out) {
  const int M = NNODE, K = DHID, Nout = DOUT;
  __shared__ __align__(16) bf16 sA[128][32];
  __shared__ __align__(16) bf16 sB[64][32];
  __shared__ float sRD[128];

  const int tid = threadIdx.x;
  const int m0 = blockIdx.x * 128;
  const int n0 = blockIdx.y * 64;

  if (tid < 128) {
    int r = m0 + tid;
    float dv = (r < M) ? deg[r] : 1.0f;
    sRD[tid] = 1.0f / fmaxf(dv, 1.0f);
  }
  __syncthreads();

  const int w = tid >> 5, lane = tid & 31;
  const int wm = (w & 3) * 32, wn = (w >> 2) * 32;
  const int half = lane >> 4, lr = lane & 15;

  const int ar  = tid >> 1;
  const int aks = (tid & 1) * 16;
  const int arow  = m0 + ar;
  const int arowc = (arow < M) ? arow : (M - 1);
  const float ascale0 = sRD[ar];
  const int bk = tid >> 3;
  const int bn = (tid & 7) * 8;
  const bool fullN = (n0 + 64 <= Nout);   // block-uniform

  v8f acc[2][2] = {};

  for (int kt = 0; kt < 2 * K; kt += 32) {
    const int phase = (kt >= K) ? 1 : 0;
    const int k0 = kt - phase * K;

    // ---- stage A: phase 0 = msg/deg (f32 -> bf16), phase 1 = h (bf16 copy) --
    if (phase == 0) {
      const float* ap = msg + (size_t)arowc * K + k0 + aks;
      float4 f[4];
      #pragma unroll
      for (int i = 0; i < 4; ++i) f[i] = ((const float4*)ap)[i];
      cvt_store16(&sA[ar][aks], f, ascale0);
    } else {
      const uint4* hp = (const uint4*)(h + (size_t)arowc * K + k0 + aks);
      uint4* q = (uint4*)&sA[ar][aks];
      q[0] = hp[0];
      q[1] = hp[1];
    }
    // ---- stage B transposed; fast path when whole 64-col tile in bounds ----
    {
      const float* W = phase ? Wr : Wl;
      if (fullN) {
        const float* wp = W + (size_t)(k0 + bk) * Nout + n0 + bn;
        float4 f0 = ((const float4*)wp)[0];
        float4 f1 = ((const float4*)wp)[1];
        bf16 hh[8];
        hh[0] = (bf16)f0.x; hh[1] = (bf16)f0.y; hh[2] = (bf16)f0.z; hh[3] = (bf16)f0.w;
        hh[4] = (bf16)f1.x; hh[5] = (bf16)f1.y; hh[6] = (bf16)f1.z; hh[7] = (bf16)f1.w;
        #pragma unroll
        for (int i = 0; i < 8; ++i) sB[bn + i][bk] = hh[i];
      } else {
        // clamped per-element path (padded cols compute garbage; never stored)
        #pragma unroll
        for (int i = 0; i < 8; ++i) {
          int col = n0 + bn + i;
          int cc  = (col < Nout) ? col : (Nout - 1);
          sB[bn + i][bk] = (bf16)W[(size_t)(k0 + bk) * Nout + cc];
        }
      }
    }
    __syncthreads();

    v16bf af[2], bfv[2];
    #pragma unroll
    for (int t = 0; t < 2; ++t) {
      af[t]  = load_afrag(&sA[wm + t * 16 + lr][0], half);
      bfv[t] = load_bfrag(&sB[wn + t * 16 + lr][0], half);
    }
    #pragma unroll
    for (int ti = 0; ti < 2; ++ti)
      #pragma unroll
      for (int tj = 0; tj < 2; ++tj)
        acc[ti][tj] = __builtin_amdgcn_wmma_f32_16x16x32_bf16(
            false, af[ti], false, bfv[tj], (short)0, acc[ti][tj], false, false);
    __syncthreads();
  }

  const bool fullM = (m0 + 128 <= M);
  #pragma unroll
  for (int ti = 0; ti < 2; ++ti) {
    #pragma unroll
    for (int tj = 0; tj < 2; ++tj) {
      int col = n0 + wn + tj * 16 + lr;
      if (col >= Nout) continue;
      float bias = b[col];
      #pragma unroll
      for (int j = 0; j < 8; ++j) {
        int row = m0 + wm + ti * 16 + j + half * 8;
        if (fullM || row < M) out[(size_t)row * Nout + col] = acc[ti][tj][j] + bias;
      }
    }
  }
}

// -------------------------- in-place log-softmax ----------------------------
__global__ void logsoftmax_k(float* __restrict__ out, int M, int C) {
  int row = (blockIdx.x * blockDim.x + threadIdx.x) >> 5;  // one wave per row
  int lane = threadIdx.x & 31;
  if (row >= M) return;
  float* p = out + (size_t)row * C;
  float mx = -__builtin_inff();
  for (int c = lane; c < C; c += 32) mx = fmaxf(mx, p[c]);
  #pragma unroll
  for (int o = 16; o > 0; o >>= 1) mx = fmaxf(mx, __shfl_xor(mx, o, 32));
  float s = 0.0f;
  for (int c = lane; c < C; c += 32) s += __expf(p[c] - mx);
  #pragma unroll
  for (int o = 16; o > 0; o >>= 1) s += __shfl_xor(s, o, 32);
  float ls = __logf(s);
  for (int c = lane; c < C; c += 32) p[c] = p[c] - mx - ls;
}

// ------------------------------- launcher -----------------------------------
extern "C" void kernel_launch(void* const* d_in, const int* in_sizes, int n_in,
                              void* d_out, int out_size, void* d_ws, size_t ws_size,
                              hipStream_t stream) {
  (void)in_sizes; (void)n_in; (void)out_size; (void)ws_size;
  const float*     x   = (const float*)d_in[0];
  const long long* ei  = (const long long*)d_in[1];   // int64 [2,E]
  const float*     W1l = (const float*)d_in[2];
  const float*     W1r = (const float*)d_in[3];
  const float*     b1  = (const float*)d_in[4];
  const float*     W2l = (const float*)d_in[5];
  const float*     W2r = (const float*)d_in[6];
  const float*     b2  = (const float*)d_in[7];
  const long long* src = ei;
  const long long* dst = ei + NEDGE;

  // workspace layout (bytes):
  //   deg : N f32                  @ 0
  //   msg : N*512 f32 (layer 1)    @ 400128   -- reused as N*256 f32 (layer 2)
  //   hbf : N*256 bf16             @ 400128 + 204800000
  char*  ws  = (char*)d_ws;
  float* deg = (float*)ws;
  float* msg = (float*)(ws + 400128);
  bf16*  hbf = (bf16*)(ws + 400128 + 204800000);
  float* out = (float*)d_out;

  zero_f32<<<2048, 256, 0, stream>>>(deg, (long)NNODE);
  zero_f32<<<4096, 256, 0, stream>>>(msg, (long)NNODE * DIN);
  degree_k<<<(NEDGE + 255) / 256, 256, 0, stream>>>(dst, deg, NEDGE);
  scatter_x<<<(int)(((long)NEDGE * 128 + 255) / 256), 256, 0, stream>>>(src, dst, x, msg, NEDGE);

  dim3 g1((NNODE + 127) / 128, DHID / 64);
  sage_gemm1<<<g1, 256, 0, stream>>>(msg, deg, x, W1l, W1r, b1, hbf);

  zero_f32<<<2048, 256, 0, stream>>>(msg, (long)NNODE * DHID);
  scatter_h<<<(int)(((long)NEDGE * 64 + 255) / 256), 256, 0, stream>>>(src, dst, hbf, msg, NEDGE);

  dim3 g2((NNODE + 127) / 128, 3);                     // 3*64 = 192 >= 129 cols
  sage_gemm2<<<g2, 256, 0, stream>>>(msg, deg, hbf, W2l, W2r, b2, out);

  logsoftmax_k<<<((NNODE * 32) + 255) / 256, 256, 0, stream>>>(out, NNODE, DOUT);
}
